// _S5Layer_34093450396044
// MI455X (gfx1250) — compile-verified, matrix operand order checked by reading, not run on hardware
//
#include <hip/hip_runtime.h>
#include <hip/hip_bf16.h>

// ---------------- problem constants ----------------
#define BB   16
#define LL   4096
#define HH   512
#define PP   256
#define NN   (BB * LL)          // 65536 tokens
#define NCHUNK 32
#define LC   (LL / NCHUNK)      // 128

typedef __bf16 bfx8  __attribute__((ext_vector_type(8)));
typedef __bf16 bfx16 __attribute__((ext_vector_type(16)));
typedef float  fx8   __attribute__((ext_vector_type(8)));

union AFrag { bfx16 v; bfx8 h[2]; };
union FAcc  { fx8  v; float f[8]; };

static __device__ __forceinline__ fx8 wmma_bf16(bfx16 a, bfx16 b, fx8 c) {
  return __builtin_amdgcn_wmma_f32_16x16x32_bf16(false, a, false, b, (short)0, c,
                                                 false, false);
}

// load a 16-element bf16 fragment: two contiguous 8-elem (16B) runs,
// base must already include row*K + kb*32 + g*8
static __device__ __forceinline__ AFrag load_frag(const __bf16* base) {
  AFrag f;
  f.h[0] = *(const bfx8*)(base);
  f.h[1] = *(const bfx8*)(base + 16);
  return f;
}

static __device__ __forceinline__ float gelu_tanh(float v) {
  return 0.5f * v * (1.0f + tanhf(0.7978845608028654f * (v + 0.044715f * v * v * v)));
}

// ---------------- 1. per-mode precompute ----------------
__global__ __launch_bounds__(256) void k_precompute(
    const float* __restrict__ log_lambda_re, const float* __restrict__ lambda_im,
    const float* __restrict__ log_step,
    float* __restrict__ lam_re, float* __restrict__ lam_im,
    float* __restrict__ abc_re, float* __restrict__ abc_im,
    float* __restrict__ cf_re,  float* __restrict__ cf_im) {
  int p = threadIdx.x;
  if (p >= PP) return;
  float step = expf(log_step[p]);
  float lre  = -expf(log_lambda_re[p]);
  float lim  = lambda_im[p];
  float ar = lre * step, ai = lim * step;
  float ea = expf(ar);
  float lbr = ea * cosf(ai), lbi = ea * sinf(ai);       // Lambda_bar
  lam_re[p] = lbr; lam_im[p] = lbi;
  float eac = expf(ar * (float)LC);
  abc_re[p] = eac * cosf(ai * (float)LC);               // Lambda_bar^LC
  abc_im[p] = eac * sinf(ai * (float)LC);
  // (Lambda_bar - 1) / Lambda
  float nr = lbr - 1.0f, ni = lbi;
  float den = lre * lre + lim * lim;
  cf_re[p] = (nr * lre + ni * lim) / den;
  cf_im[p] = (ni * lre - nr * lim) / den;
}

// ---------------- 2. B_bar = coef * B_tilde  (bf16, [P,H] row-major) ------
__global__ __launch_bounds__(256) void k_bbar(
    const float* __restrict__ B_re, const float* __restrict__ B_im,
    const float* __restrict__ cf_re, const float* __restrict__ cf_im,
    __bf16* __restrict__ bbar_re, __bf16* __restrict__ bbar_im) {
  int idx = blockIdx.x * 256 + threadIdx.x;   // idx over P*H
  int p = idx >> 9;                            // /H
  float cr = cf_re[p], ci = cf_im[p];
  float br = B_re[idx], bi = B_im[idx];
  bbar_re[idx] = (__bf16)(cr * br - ci * bi);
  bbar_im[idx] = (__bf16)(cr * bi + ci * br);
}

// ---------------- 3. C scaled copies (fold 2*Re(.) and the minus sign) ----
__global__ __launch_bounds__(256) void k_cconv(
    const float* __restrict__ C_re, const float* __restrict__ C_im,
    __bf16* __restrict__ c2_re, __bf16* __restrict__ c2_im) {
  int idx = blockIdx.x * 256 + threadIdx.x;   // idx over H*P
  c2_re[idx] = (__bf16)( 2.0f * C_re[idx]);
  c2_im[idx] = (__bf16)(-2.0f * C_im[idx]);
}

// ---------------- 4. GEMM1: Bu = x @ B_bar^T (re & im) --------------------
__global__ __launch_bounds__(256) void k_gemm1(
    const float* __restrict__ x,
    const __bf16* __restrict__ bbar_re, const __bf16* __restrict__ bbar_im,
    float* __restrict__ bu_re, float* __restrict__ bu_im) {
  __shared__ __align__(16) __bf16 sx[16 * HH];          // 16 KB bf16 x-tile
  const int tid = threadIdx.x;
  const size_t m0 = (size_t)blockIdx.x * 16;
  const float4* xp = (const float4*)(x + m0 * HH);      // tile is contiguous
  for (int i = tid; i < 16 * HH / 4; i += 256) {
    float4 v = xp[i];
    sx[4 * i + 0] = (__bf16)v.x;
    sx[4 * i + 1] = (__bf16)v.y;
    sx[4 * i + 2] = (__bf16)v.z;
    sx[4 * i + 3] = (__bf16)v.w;
  }
  __syncthreads();
  const int lane = tid & 31, wave = tid >> 5;
  const int g = lane >> 4, rl = lane & 15;
  for (int wt = 0; wt < 2; ++wt) {
    const int n0 = (wave * 2 + wt) * 16;                // P-tile origin
    FAcc ar = {}, ai = {};
    for (int kb = 0; kb < HH / 32; ++kb) {
      AFrag a  = load_frag(&sx[rl * HH + kb * 32 + g * 8]);
      AFrag br = load_frag(bbar_re + (size_t)(n0 + rl) * HH + kb * 32 + g * 8);
      AFrag bi = load_frag(bbar_im + (size_t)(n0 + rl) * HH + kb * 32 + g * 8);
      ar.v = wmma_bf16(a.v, br.v, ar.v);
      ai.v = wmma_bf16(a.v, bi.v, ai.v);
    }
    for (int r = 0; r < 8; ++r) {
      size_t m = m0 + r + 8 * g;
      bu_re[m * PP + n0 + rl] = ar.f[r];
      bu_im[m * PP + n0 + rl] = ai.f[r];
    }
  }
}

// ---------------- 5. scan phase 1: chunk-local prefix (in place) ----------
__global__ __launch_bounds__(256) void k_scan1(
    float* __restrict__ bu_re, float* __restrict__ bu_im,
    const float* __restrict__ lam_re, const float* __restrict__ lam_im,
    float* __restrict__ carry_re, float* __restrict__ carry_im) {
  int idx = blockIdx.x * 256 + threadIdx.x;
  int p = idx & (PP - 1);
  int c = (idx >> 8) & (NCHUNK - 1);
  int b = idx >> 13;
  float lr = lam_re[p], li = lam_im[p];
  size_t base = ((size_t)(b * LL + c * LC)) * PP + p;
  float sr = 0.0f, si = 0.0f;
  for (int j = 0; j < LC; ++j) {
    float ur = bu_re[base], ui = bu_im[base];
    float nr = lr * sr - li * si + ur;
    float ni = lr * si + li * sr + ui;
    sr = nr; si = ni;
    bu_re[base] = sr; bu_im[base] = si;
    base += PP;
  }
  carry_re[(size_t)(b * NCHUNK + c) * PP + p] = sr;
  carry_im[(size_t)(b * NCHUNK + c) * PP + p] = si;
}

// ---------------- 6. scan phase 2: cross-chunk combine --------------------
__global__ __launch_bounds__(256) void k_scan2(
    const float* __restrict__ carry_re, const float* __restrict__ carry_im,
    const float* __restrict__ abc_re, const float* __restrict__ abc_im,
    float* __restrict__ cin_re, float* __restrict__ cin_im) {
  int idx = blockIdx.x * 256 + threadIdx.x;   // B*P threads
  int p = idx & (PP - 1);
  int b = idx >> 8;
  float ar = abc_re[p], ai = abc_im[p];
  float sr = 0.0f, si = 0.0f;
  for (int c = 0; c < NCHUNK; ++c) {
    size_t o = (size_t)(b * NCHUNK + c) * PP + p;
    cin_re[o] = sr; cin_im[o] = si;
    float tr = carry_re[o], ti = carry_im[o];
    float nr = ar * sr - ai * si + tr;
    float ni = ar * si + ai * sr + ti;
    sr = nr; si = ni;
  }
}

// ---------------- 7. scan phase 3: fixup, emit xs as bf16 -----------------
__global__ __launch_bounds__(256) void k_scan3(
    const float* __restrict__ bu_re, const float* __restrict__ bu_im,
    const float* __restrict__ lam_re, const float* __restrict__ lam_im,
    const float* __restrict__ cin_re, const float* __restrict__ cin_im,
    __bf16* __restrict__ xs_re, __bf16* __restrict__ xs_im) {
  int idx = blockIdx.x * 256 + threadIdx.x;
  int p = idx & (PP - 1);
  int c = (idx >> 8) & (NCHUNK - 1);
  int b = idx >> 13;
  float lr = lam_re[p], li = lam_im[p];
  size_t co = (size_t)(b * NCHUNK + c) * PP + p;
  float sr = cin_re[co], si = cin_im[co];
  float fr = lr, fi = li;                      // Lambda_bar^(j+1)
  size_t base = ((size_t)(b * LL + c * LC)) * PP + p;
  for (int j = 0; j < LC; ++j) {
    float vr = bu_re[base] + fr * sr - fi * si;
    float vi = bu_im[base] + fr * si + fi * sr;
    xs_re[base] = (__bf16)vr;
    xs_im[base] = (__bf16)vi;
    float nfr = fr * lr - fi * li;
    float nfi = fr * li + fi * lr;
    fr = nfr; fi = nfi;
    base += PP;
  }
}

// ---------------- 8. GEMM2 + D-skip + GELU + residual + LayerNorm ---------
__global__ __launch_bounds__(256) void k_gemm2_ln(
    const __bf16* __restrict__ xs_re, const __bf16* __restrict__ xs_im,
    const __bf16* __restrict__ c2_re, const __bf16* __restrict__ c2_im,
    const float* __restrict__ x, const float* __restrict__ Dv,
    const float* __restrict__ lns, const float* __restrict__ lnb,
    float* __restrict__ out) {
  __shared__ __align__(16) __bf16 sre[16 * PP];         // 8 KB
  __shared__ __align__(16) __bf16 sim[16 * PP];         // 8 KB
  __shared__ __align__(16) float  sy[16 * HH];          // 32 KB ys tile
  const int tid = threadIdx.x;
  const size_t m0 = (size_t)blockIdx.x * 16;
  // stage xs tiles (contiguous 16*P blocks) into LDS
  const bfx8* gr = (const bfx8*)(xs_re + m0 * PP);
  const bfx8* gi = (const bfx8*)(xs_im + m0 * PP);
  for (int i = tid; i < 16 * PP / 8; i += 256) {
    ((bfx8*)sre)[i] = gr[i];
    ((bfx8*)sim)[i] = gi[i];
  }
  __syncthreads();
  const int lane = tid & 31, wave = tid >> 5;
  const int g = lane >> 4, rl = lane & 15;
  for (int wt = 0; wt < 4; ++wt) {
    const int n0 = (wave * 4 + wt) * 16;                // H-tile origin
    FAcc acc = {};
    for (int kb = 0; kb < PP / 32; ++kb) {
      AFrag are = load_frag(&sre[rl * PP + kb * 32 + g * 8]);
      AFrag aim = load_frag(&sim[rl * PP + kb * 32 + g * 8]);
      AFrag bre = load_frag(c2_re + (size_t)(n0 + rl) * PP + kb * 32 + g * 8);
      AFrag bim = load_frag(c2_im + (size_t)(n0 + rl) * PP + kb * 32 + g * 8);
      acc.v = wmma_bf16(are.v, bre.v, acc.v);           // + 2*xs_re*C_re
      acc.v = wmma_bf16(aim.v, bim.v, acc.v);           // - 2*xs_im*C_im
    }
    for (int r = 0; r < 8; ++r)
      sy[(r + 8 * g) * HH + n0 + rl] = acc.f[r];
  }
  __syncthreads();
  // LayerNorm: 16 threads per token row, 32 columns each
  const int row = tid >> 4, cx = tid & 15;
  const float* xrow = x + (m0 + row) * HH;
  float sum = 0.0f, sumsq = 0.0f;
  for (int j = 0; j < HH / 16; ++j) {
    int col = cx + j * 16;
    float xv = xrow[col];
    float yv = sy[row * HH + col] + xv * Dv[col];
    float z  = xv + gelu_tanh(yv);
    sy[row * HH + col] = z;
    sum += z; sumsq += z * z;
  }
  for (int off = 1; off < 16; off <<= 1) {
    sum   += __shfl_xor(sum, off, 32);
    sumsq += __shfl_xor(sumsq, off, 32);
  }
  float mean = sum * (1.0f / HH);
  float var  = sumsq * (1.0f / HH) - mean * mean;
  float rstd = rsqrtf(var + 1e-6f);
  float* orow = out + (m0 + row) * HH;
  for (int j = 0; j < HH / 16; ++j) {
    int col = cx + j * 16;
    float z = sy[row * HH + col];
    orow[col] = (z - mean) * rstd * lns[col] + lnb[col];
  }
}

// ---------------- host launcher ----------------
extern "C" void kernel_launch(void* const* d_in, const int* in_sizes, int n_in,
                              void* d_out, int out_size, void* d_ws, size_t ws_size,
                              hipStream_t stream) {
  (void)in_sizes; (void)n_in; (void)out_size; (void)ws_size;
  const float* x      = (const float*)d_in[0];
  const float* llre   = (const float*)d_in[1];
  const float* lim    = (const float*)d_in[2];
  const float* B_re   = (const float*)d_in[3];
  const float* B_im   = (const float*)d_in[4];
  const float* C_re   = (const float*)d_in[5];
  const float* C_im   = (const float*)d_in[6];
  const float* Dv     = (const float*)d_in[7];
  const float* lstep  = (const float*)d_in[8];
  const float* lns    = (const float*)d_in[9];
  const float* lnb    = (const float*)d_in[10];
  float* out = (float*)d_out;

  // workspace carve-up
  const size_t NP = (size_t)NN * PP;
  const size_t PH = (size_t)PP * HH;
  const size_t BC = (size_t)BB * NCHUNK * PP;
  char* w = (char*)d_ws;
  float*  bu_re   = (float*)w;            w += NP * sizeof(float);
  float*  bu_im   = (float*)w;            w += NP * sizeof(float);
  __bf16* xs_re   = (__bf16*)w;           w += NP * sizeof(__bf16);
  __bf16* xs_im   = (__bf16*)w;           w += NP * sizeof(__bf16);
  __bf16* bbar_re = (__bf16*)w;           w += PH * sizeof(__bf16);
  __bf16* bbar_im = (__bf16*)w;           w += PH * sizeof(__bf16);
  __bf16* c2_re   = (__bf16*)w;           w += PH * sizeof(__bf16);
  __bf16* c2_im   = (__bf16*)w;           w += PH * sizeof(__bf16);
  float*  lam_re  = (float*)w;            w += PP * sizeof(float);
  float*  lam_im  = (float*)w;            w += PP * sizeof(float);
  float*  abc_re  = (float*)w;            w += PP * sizeof(float);
  float*  abc_im  = (float*)w;            w += PP * sizeof(float);
  float*  cf_re   = (float*)w;            w += PP * sizeof(float);
  float*  cf_im   = (float*)w;            w += PP * sizeof(float);
  float*  carry_re= (float*)w;            w += BC * sizeof(float);
  float*  carry_im= (float*)w;            w += BC * sizeof(float);
  float*  cin_re  = (float*)w;            w += BC * sizeof(float);
  float*  cin_im  = (float*)w;            w += BC * sizeof(float);

  k_precompute<<<1, 256, 0, stream>>>(llre, lim, lstep,
                                      lam_re, lam_im, abc_re, abc_im, cf_re, cf_im);
  k_bbar<<<(PP * HH) / 256, 256, 0, stream>>>(B_re, B_im, cf_re, cf_im,
                                              bbar_re, bbar_im);
  k_cconv<<<(HH * PP) / 256, 256, 0, stream>>>(C_re, C_im, c2_re, c2_im);
  k_gemm1<<<NN / 16, 256, 0, stream>>>(x, bbar_re, bbar_im, bu_re, bu_im);
  k_scan1<<<(BB * NCHUNK * PP) / 256, 256, 0, stream>>>(bu_re, bu_im,
                                                        lam_re, lam_im,
                                                        carry_re, carry_im);
  k_scan2<<<(BB * PP) / 256, 256, 0, stream>>>(carry_re, carry_im,
                                               abc_re, abc_im, cin_re, cin_im);
  k_scan3<<<(BB * NCHUNK * PP) / 256, 256, 0, stream>>>(bu_re, bu_im,
                                                        lam_re, lam_im,
                                                        cin_re, cin_im,
                                                        xs_re, xs_im);
  k_gemm2_ln<<<NN / 16, 256, 0, stream>>>(xs_re, xs_im, c2_re, c2_im,
                                          x, Dv, lns, lnb, out);
}
